// TaskGraphMatcher_50929722196067
// MI455X (gfx1250) — compile-verified, matrix operand order checked by reading, not run on hardware
//
#include <hip/hip_runtime.h>
#include <math.h>

typedef __attribute__((ext_vector_type(16))) _Float16 v16h;
typedef __attribute__((ext_vector_type(8)))  float    v8f;

#define B_   32
#define S_   400
#define N_   64
#define D_   256
#define H_   4
#define DH_  64
#define L_   2
#define DFF_ 1024
#define VD_  768
#define TD_  512

// gfx1250 async global->LDS copy path (ASYNCcnt-tracked), guarded so the file
// compiles on toolchains (and the host pass) where the builtins are absent.
#if __has_builtin(__builtin_amdgcn_global_load_async_to_lds_b32) && \
    __has_builtin(__builtin_amdgcn_s_wait_asynccnt)
#define HAVE_ASYNC_LDS 1
typedef __attribute__((address_space(1))) int* as1_i32p;
typedef __attribute__((address_space(3))) int* as3_i32p;
#else
#define HAVE_ASYNC_LDS 0
#endif

// ---------------------------------------------------------------- WMMA helpers

__device__ __forceinline__ v8f wmma_f16(v16h a, v16h b, v8f c) {
  // D = A(16x32 f16) * B(32x16 f16) + C(16x16 f32)
  return __builtin_amdgcn_wmma_f32_16x16x32_f16(false, a, false, b, (short)0, c,
                                                false, false);
}

// A-fragment (16x32, MxK) from fp32 row-major [rows, ld].
// ISA 7.12.2: lane<16 -> row=lane, k = k0+0..7 / k0+16..23
//             lane>=16 -> row=lane-16, k = k0+8..15 / k0+24..31
__device__ __forceinline__ v16h load_frag_rm(const float* __restrict__ base, int ld,
                                             int row0, int k0, int lane) {
  int r  = row0 + (lane & 15);
  int kb = k0 + ((lane >> 4) << 3);
  const float* p = base + (size_t)r * ld + kb;
  v16h f;
#pragma unroll
  for (int i = 0; i < 8; ++i) f[i] = (_Float16)p[i];
#pragma unroll
  for (int i = 0; i < 8; ++i) f[8 + i] = (_Float16)p[16 + i];
  return f;
}

// B-fragment (32x16, KxN) from fp32 stored k-major [K, ld] (element [k][n]),
// with k bound (zero pad beyond klim).
__device__ __forceinline__ v16h load_frag_km(const float* __restrict__ base, int ld,
                                             int k0, int n0, int klim, int lane) {
  int n  = n0 + (lane & 15);
  int kb = k0 + ((lane >> 4) << 3);
  v16h f;
#pragma unroll
  for (int i = 0; i < 8; ++i) {
    int k = kb + i;
    f[i] = (k < klim) ? (_Float16)base[(size_t)k * ld + n] : (_Float16)0.f;
  }
#pragma unroll
  for (int i = 0; i < 8; ++i) {
    int k = kb + 16 + i;
    f[8 + i] = (k < klim) ? (_Float16)base[(size_t)k * ld + n] : (_Float16)0.f;
  }
  return f;
}

__device__ __forceinline__ float wave_sum(float v) {
#pragma unroll
  for (int off = 16; off > 0; off >>= 1) v += __shfl_down(v, off);
  return __shfl(v, 0);
}

// ---------------------------------------------------------------- generic GEMM
// C[M,N] = A[M,K] @ W[N,K]^T + bias, optional relu.
// 4-wave workgroup -> 64x64 output tile; the shared 64x32 fp32 W panel is
// staged through LDS once per k-step (async copy on gfx1250 when available),
// giving 4x weight reuse; each wave owns 16 rows and reads A directly.
__global__ __launch_bounds__(128)
void gemm_wmma_kernel(const float* __restrict__ A, const float* __restrict__ W,
                      const float* __restrict__ bias, float* __restrict__ C,
                      int K, int lda, int ldw, int ldc, int relu) {
  __shared__ float wt[64][32];
  int tid  = threadIdx.x;
  int lane = tid & 31;
  int wave = tid >> 5;
  int row0 = blockIdx.y * 64 + wave * 16;
  int col0 = blockIdx.x * 64;
  v8f acc[4] = {};
  for (int k0 = 0; k0 < K; k0 += 32) {
    __syncthreads();
#if HAVE_ASYNC_LDS
    for (int e = tid; e < 2048; e += 128) {
      int col = e >> 5, k = e & 31;
      const float* g = W + (size_t)(col0 + col) * ldw + (k0 + k);
      __builtin_amdgcn_global_load_async_to_lds_b32(
          (as1_i32p)(void*)(size_t)(const void*)g,
          (as3_i32p)(void*)&wt[col][k], 0, 0);
    }
    __builtin_amdgcn_s_wait_asynccnt(0);
#else
    for (int e = tid; e < 2048; e += 128) {
      int col = e >> 5, k = e & 31;
      wt[col][k] = W[(size_t)(col0 + col) * ldw + (k0 + k)];
    }
#endif
    __syncthreads();
    if (k0 + 32 < K)  // prefetch next A panel (lowers to global_prefetch_b8)
      __builtin_prefetch(A + (size_t)(row0 + (lane & 15)) * lda + (k0 + 32), 0, 1);
    v16h a = load_frag_rm(A, lda, row0, k0, lane);
#pragma unroll
    for (int t = 0; t < 4; ++t) {
      int n  = t * 16 + (lane & 15);
      int kb = (lane >> 4) << 3;
      v16h bf;
#pragma unroll
      for (int i = 0; i < 8; ++i) bf[i] = (_Float16)wt[n][kb + i];
#pragma unroll
      for (int i = 0; i < 8; ++i) bf[8 + i] = (_Float16)wt[n][kb + 16 + i];
      acc[t] = wmma_f16(a, bf, acc[t]);
    }
  }
  int ccol = lane & 15;
  int rb   = row0 + ((lane >> 4) ? 8 : 0);
#pragma unroll
  for (int t = 0; t < 4; ++t) {
    int col  = col0 + 16 * t + ccol;
    float bb = bias ? bias[col] : 0.f;
#pragma unroll
    for (int r = 0; r < 8; ++r) {
      float v = acc[t][r] + bb;
      if (relu) v = fmaxf(v, 0.f);
      C[(size_t)(rb + r) * ldc + col] = v;
    }
  }
}

// ---------------------------------------------------------------- LayerNorm
// out[row] = [relu?] LN(x[row] (+ res[row])) * g + b  (+ pos[row % S])
__global__ __launch_bounds__(32)
void ln_kernel(const float* __restrict__ x, const float* __restrict__ res,
               const float* __restrict__ g, const float* __restrict__ bta,
               const float* __restrict__ pos, float* __restrict__ out, int relu) {
  int row  = blockIdx.x;
  int lane = threadIdx.x;
  const float* xr = x + (size_t)row * D_;
  const float* rr = res ? res + (size_t)row * D_ : nullptr;
  float vals[8];
  float ls = 0.f;
#pragma unroll
  for (int i = 0; i < 8; ++i) {
    int c   = lane * 8 + i;
    float t = xr[c] + (rr ? rr[c] : 0.f);
    vals[i] = t;
    ls += t;
  }
  float mu = wave_sum(ls) * (1.f / D_);
  float lv = 0.f;
#pragma unroll
  for (int i = 0; i < 8; ++i) { float d = vals[i] - mu; lv += d * d; }
  float rstd = rsqrtf(wave_sum(lv) * (1.f / D_) + 1e-5f);
#pragma unroll
  for (int i = 0; i < 8; ++i) {
    int c   = lane * 8 + i;
    float y = (vals[i] - mu) * rstd * g[c] + bta[c];
    if (relu) y = fmaxf(y, 0.f);
    if (pos) y += pos[(size_t)(row % S_) * D_ + c];
    out[(size_t)row * D_ + c] = y;
  }
}

// ---------------------------------------------------------------- attention
// One wave per (b, h, 16-query tile). Scores [16 x 400] kept in LDS; softmax
// in LDS; P@V back through WMMA. qkv layout [B,S,3D], out [B,S,D].
__global__ __launch_bounds__(32)
void attn_kernel(const float* __restrict__ qkv, float* __restrict__ out) {
  __shared__ float sc[16][416];
  __shared__ float rowsum[16];
  int lane = threadIdx.x;
  int q0   = blockIdx.x * 16;
  int h    = blockIdx.y;
  int b    = blockIdx.z;
  const float* base = qkv + (size_t)b * S_ * (3 * D_);
  const float* Qb = base + h * DH_;
  const float* Kb = base + D_ + h * DH_;
  const float* Vb = base + 2 * D_ + h * DH_;

  v16h qa0 = load_frag_rm(Qb, 3 * D_, q0, 0, lane);
  v16h qa1 = load_frag_rm(Qb, 3 * D_, q0, 32, lane);
  int ccol = lane & 15;
  int rb   = (lane >> 4) ? 8 : 0;

  for (int kt = 0; kt < S_ / 16; ++kt) {        // 25 key tiles of 16
    v16h kb0 = load_frag_rm(Kb, 3 * D_, kt * 16, 0, lane);
    v16h kb1 = load_frag_rm(Kb, 3 * D_, kt * 16, 32, lane);
    v8f s = {};
    s = wmma_f16(qa0, kb0, s);
    s = wmma_f16(qa1, kb1, s);
#pragma unroll
    for (int r = 0; r < 8; ++r) sc[rb + r][kt * 16 + ccol] = s[r] * 0.125f; // 1/sqrt(64)
  }
  __syncthreads();

  if (lane < 16) {                               // row-wise softmax (unnormalized)
    float m = -1e30f;
    for (int j = 0; j < S_; ++j) m = fmaxf(m, sc[lane][j]);
    float ssum = 0.f;
    for (int j = 0; j < S_; ++j) { float e = __expf(sc[lane][j] - m); sc[lane][j] = e; ssum += e; }
    for (int j = S_; j < 416; ++j) sc[lane][j] = 0.f;
    rowsum[lane] = ssum;
  }
  __syncthreads();

  v8f o[4] = {};
  for (int kt = 0; kt < 13; ++kt) {              // 13 key tiles of 32 (padded 416)
    int kk = kt * 32;
    v16h pa;                                     // A-frag of P from LDS
    {
      int r  = lane & 15;
      int kb = kk + ((lane >> 4) << 3);
#pragma unroll
      for (int i = 0; i < 8; ++i) pa[i] = (_Float16)sc[r][kb + i];
#pragma unroll
      for (int i = 0; i < 8; ++i) pa[8 + i] = (_Float16)sc[r][kb + 16 + i];
    }
#pragma unroll
    for (int t = 0; t < 4; ++t) {
      v16h vb = load_frag_km(Vb, 3 * D_, kk, t * 16, S_, lane);
      o[t] = wmma_f16(pa, vb, o[t]);
    }
  }
#pragma unroll
  for (int t = 0; t < 4; ++t)
#pragma unroll
    for (int r = 0; r < 8; ++r) {
      float rs = 1.f / rowsum[rb + r];
      out[(size_t)b * S_ * D_ + (size_t)(q0 + rb + r) * D_ + h * DH_ + t * 16 + ccol] =
          o[t][r] * rs;
    }
}

// ---------------------------------------------------------------- row rnorm
__global__ __launch_bounds__(32)
void rownorm_kernel(const float* __restrict__ x, float* __restrict__ rn) {
  int row = blockIdx.x, lane = threadIdx.x;
  const float* xr = x + (size_t)row * D_;
  float s = 0.f;
#pragma unroll
  for (int i = 0; i < 8; ++i) { float t = xr[lane * 8 + i]; s += t * t; }
  s = wave_sum(s);
  if (lane == 0) rn[row] = 1.f / fmaxf(sqrtf(s), 1e-12f);
}

// ---------------------------------------------------------------- sim = vn @ tn^T
__global__ __launch_bounds__(32)
void sim_kernel(const float* __restrict__ vis, const float* __restrict__ txt,
                const float* __restrict__ rv, const float* __restrict__ rt,
                float* __restrict__ sim) {
  int lane = threadIdx.x;
  int s0   = blockIdx.x * 16;
  int b    = blockIdx.y;
  const float* A  = vis + (size_t)b * S_ * D_;
  const float* Bm = txt + (size_t)b * N_ * D_;
  v8f acc[4] = {};
  for (int k0 = 0; k0 < D_; k0 += 32) {
    v16h a = load_frag_rm(A, D_, s0, k0, lane);
#pragma unroll
    for (int t = 0; t < 4; ++t) {
      v16h bf = load_frag_rm(Bm, D_, t * 16, k0, lane);
      acc[t] = wmma_f16(a, bf, acc[t]);
    }
  }
  int ccol = lane & 15, rb = (lane >> 4) ? 8 : 0;
#pragma unroll
  for (int t = 0; t < 4; ++t)
#pragma unroll
    for (int r = 0; r < 8; ++r) {
      int s = s0 + rb + r, n = t * 16 + ccol;
      sim[(size_t)b * S_ * N_ + (size_t)s * N_ + n] =
          acc[t][r] * rv[b * S_ + s] * rt[b * N_ + n];
    }
}

// ---------------------------------------------------------------- Hungarian LAP
// One wave per batch. cost[row=n][col=s] = -sim[b][s][n]; rows=64 <= cols=400.
__global__ __launch_bounds__(32)
void lap_kernel(const float* __restrict__ sim, int* __restrict__ steps) {
  __shared__ float u[N_ + 1], v[S_ + 1], minv[S_ + 1];
  __shared__ int p[S_ + 1], way[S_ + 1], used[S_ + 1];
  __shared__ float s_delta;
  __shared__ int s_j1, s_j0;
  const float INF = 1e30f;
  int b = blockIdx.x, tid = threadIdx.x;
  const float* sb = sim + (size_t)b * S_ * N_;

  for (int j = tid; j <= S_; j += 32) { v[j] = 0.f; p[j] = 0; way[j] = 0; }
  for (int i = tid; i <= N_; i += 32) u[i] = 0.f;
  __syncthreads();

  for (int i = 1; i <= N_; ++i) {
    if (tid == 0) { p[0] = i; s_j0 = 0; }
    for (int j = tid; j <= S_; j += 32) { minv[j] = INF; used[j] = 0; }
    __syncthreads();
    while (true) {
      int j0 = s_j0;
      if (tid == 0) used[j0] = 1;
      __syncthreads();
      int i0    = p[j0];
      float ui0 = u[i0];
      float best = INF;
      int bj = -1;
      for (int j = tid + 1; j <= S_; j += 32) {
        if (!used[j]) {
          float cur = -sb[(size_t)(j - 1) * N_ + (i0 - 1)] - ui0 - v[j];
          if (cur < minv[j]) { minv[j] = cur; way[j] = j0; }
          float mv = minv[j];
          if (mv < best) { best = mv; bj = j; }
        }
      }
#pragma unroll
      for (int off = 16; off > 0; off >>= 1) {
        float ob = __shfl_down(best, off);
        int oj   = __shfl_down(bj, off);
        if (ob < best) { best = ob; bj = oj; }
      }
      if (tid == 0) { s_delta = best; s_j1 = bj; }
      __syncthreads();
      float delta = s_delta;
      int j1 = s_j1;
      for (int j = tid; j <= S_; j += 32) {
        if (used[j]) { u[p[j]] += delta; v[j] -= delta; }
        else minv[j] -= delta;
      }
      __syncthreads();
      if (tid == 0) s_j0 = j1;
      __syncthreads();
      if (p[j1] == 0) break;
    }
    if (tid == 0) {                 // augment along the alternating path
      int j0 = s_j0;
      while (j0) { int j1 = way[j0]; p[j0] = p[j1]; j0 = j1; }
    }
    __syncthreads();
  }
  for (int j = tid + 1; j <= S_; j += 32) {
    int r = p[j];
    if (r) steps[b * N_ + r - 1] = j - 1;
  }
}

// ---------------------------------------------------------------- head pieces
__global__ __launch_bounds__(64)
void gather_cost_kernel(const float* __restrict__ sim, const int* __restrict__ steps,
                        float* __restrict__ sim_m, float* __restrict__ cost_out) {
  __shared__ float red[64];
  int b = blockIdx.x, n = threadIdx.x;
  int s = steps[b * N_ + n];
  float sm = sim[(size_t)b * S_ * N_ + (size_t)s * N_ + n];
  sim_m[b * N_ + n] = sm;
  red[n] = sm;
  __syncthreads();
  for (int off = 32; off > 0; off >>= 1) {
    if (n < off) red[n] += red[n + off];
    __syncthreads();
  }
  if (n == 0) cost_out[b] = red[0] * (1.f / N_);
}

// pre[b, steps[b,n], d] += txt[b,n] . fus_w[d,256:512] + sim_m[b,n]*fus_w[d,512]
__global__ __launch_bounds__(256)
void fus_add_kernel(const float* __restrict__ txt, const float* __restrict__ fus_w,
                    const float* __restrict__ sim_m, const int* __restrict__ steps,
                    float* __restrict__ pre) {
  int n = blockIdx.x, b = blockIdx.y, d = threadIdx.x;
  int s = steps[b * N_ + n];
  const float* tr = txt + (size_t)(b * N_ + n) * D_;
  const float* wr = fus_w + (size_t)d * 513 + D_;
  float acc = 0.f;
  for (int k = 0; k < D_; ++k) acc += tr[k] * wr[k];
  acc += sim_m[b * N_ + n] * fus_w[(size_t)d * 513 + 512];
  pre[(size_t)b * S_ * D_ + (size_t)s * D_ + d] += acc;
}

__global__ __launch_bounds__(256)
void pool_kernel(const float* __restrict__ pre, float* __restrict__ pooled) {
  int b = blockIdx.x, d = threadIdx.x;
  float acc = 0.f;
  const float* pb = pre + (size_t)b * S_ * D_ + d;
  for (int s = 0; s < S_; ++s) acc += fmaxf(pb[(size_t)s * D_], 0.f);
  pooled[b * D_ + d] = acc * (1.f / S_);
}

__global__ __launch_bounds__(64)
void cls_kernel(const float* __restrict__ pooled, const float* __restrict__ w1,
                const float* __restrict__ b1, const float* __restrict__ w2,
                const float* __restrict__ b2, float* __restrict__ prob) {
  __shared__ float hid[64];
  int b = blockIdx.x, j = threadIdx.x;
  const float* pb = pooled + b * D_;
  float acc = b1[j];
  for (int k = 0; k < D_; ++k) acc += pb[k] * w1[j * D_ + k];
  hid[j] = fmaxf(acc, 0.f);
  __syncthreads();
  if (j == 0) {
    float z = b2[0];
    for (int k = 0; k < 64; ++k) z += hid[k] * w2[k];
    prob[b] = 1.f / (1.f + __expf(-z));
  }
}

// ---------------------------------------------------------------- launch
extern "C" void kernel_launch(void* const* d_in, const int* in_sizes, int n_in,
                              void* d_out, int out_size, void* d_ws, size_t ws_size,
                              hipStream_t stream) {
  const float* visual  = (const float*)d_in[0];
  const float* text    = (const float*)d_in[1];
  const float* vproj_w = (const float*)d_in[2];
  const float* vproj_b = (const float*)d_in[3];
  const float* vln_g   = (const float*)d_in[4];
  const float* vln_b   = (const float*)d_in[5];
  const float* tproj_w = (const float*)d_in[6];
  const float* tproj_b = (const float*)d_in[7];
  const float* tln_g   = (const float*)d_in[8];
  const float* tln_b   = (const float*)d_in[9];
  const float* pos     = (const float*)d_in[10];
  const float* qkv_w   = (const float*)d_in[11];
  const float* qkv_b   = (const float*)d_in[12];
  const float* out_w   = (const float*)d_in[13];
  const float* out_b   = (const float*)d_in[14];
  const float* ln1_g   = (const float*)d_in[15];
  const float* ln1_b   = (const float*)d_in[16];
  const float* ff1_w   = (const float*)d_in[17];
  const float* ff1_b   = (const float*)d_in[18];
  const float* ff2_w   = (const float*)d_in[19];
  const float* ff2_b   = (const float*)d_in[20];
  const float* ln2_g   = (const float*)d_in[21];
  const float* ln2_b   = (const float*)d_in[22];
  // d_in[23..26]: cross-attention params, dead code in reference
  const float* fus_w   = (const float*)d_in[27];
  const float* fus_b   = (const float*)d_in[28];
  const float* cls1_w  = (const float*)d_in[29];
  const float* cls1_b  = (const float*)d_in[30];
  const float* cls2_w  = (const float*)d_in[31];
  const float* cls2_b  = (const float*)d_in[32];
  float* outp = (float*)d_out;           // [0..31]=prob, [32..63]=matching cost

  const int MROWS = B_ * S_;             // 12800
  const int TROWS = B_ * N_;             // 2048
  float* w = (float*)d_ws;
  size_t off = 0;
  float* vis  = w + off; off += (size_t)MROWS * D_;        // 3.28M
  float* txt  = w + off; off += (size_t)TROWS * D_;        // 0.52M
  float* R    = w + off; off += (size_t)MROWS * 4 * D_;    // 13.1M (qkv/attn | hidden | pre)
  float* bo   = w + off; off += (size_t)MROWS * D_;        // 3.28M
  float* simb = w + off; off += (size_t)B_ * S_ * N_;      // 0.82M
  float* rv   = w + off; off += MROWS;
  float* rt   = w + off; off += TROWS;
  float* simm = w + off; off += TROWS;
  float* pooled = w + off; off += B_ * D_;
  int* steps  = (int*)(w + off);
  float* qkv  = R;                       // [B,S,3D]
  float* aout = R + (size_t)MROWS * 3 * D_;  // [B,S,D]
  float* hid  = R;                       // [B,S,DFF]
  float* pre  = R;                       // [B,S,D]
  (void)ws_size; (void)n_in; (void)in_sizes; (void)out_size;

  dim3 w32(32), w128(128);

  // visual projection -> LN -> relu -> +pos
  gemm_wmma_kernel<<<dim3(D_ / 64, MROWS / 64), w128, 0, stream>>>(
      visual, vproj_w, vproj_b, bo, VD_, VD_, VD_, D_, 0);
  ln_kernel<<<dim3(MROWS), w32, 0, stream>>>(bo, nullptr, vln_g, vln_b, pos, vis, 1);

  // text projection -> LN -> relu
  gemm_wmma_kernel<<<dim3(D_ / 64, TROWS / 64), w128, 0, stream>>>(
      text, tproj_w, tproj_b, bo, TD_, TD_, TD_, D_, 0);
  ln_kernel<<<dim3(TROWS), w32, 0, stream>>>(bo, nullptr, tln_g, tln_b, nullptr, txt, 1);

  // encoder layers (post-norm)
  for (int l = 0; l < L_; ++l) {
    gemm_wmma_kernel<<<dim3(3 * D_ / 64, MROWS / 64), w128, 0, stream>>>(
        vis, qkv_w + (size_t)l * 3 * D_ * D_, qkv_b + l * 3 * D_, qkv,
        D_, D_, D_, 3 * D_, 0);
    attn_kernel<<<dim3(S_ / 16, H_, B_), w32, 0, stream>>>(qkv, aout);
    gemm_wmma_kernel<<<dim3(D_ / 64, MROWS / 64), w128, 0, stream>>>(
        aout, out_w + (size_t)l * D_ * D_, out_b + l * D_, R, D_, D_, D_, D_, 0);
    ln_kernel<<<dim3(MROWS), w32, 0, stream>>>(
        R, vis, ln1_g + l * D_, ln1_b + l * D_, nullptr, vis, 0);
    gemm_wmma_kernel<<<dim3(DFF_ / 64, MROWS / 64), w128, 0, stream>>>(
        vis, ff1_w + (size_t)l * DFF_ * D_, ff1_b + l * DFF_, hid, D_, D_, D_, DFF_, 1);
    gemm_wmma_kernel<<<dim3(D_ / 64, MROWS / 64), w128, 0, stream>>>(
        hid, ff2_w + (size_t)l * D_ * DFF_, ff2_b + l * D_, bo, DFF_, DFF_, DFF_, D_, 0);
    ln_kernel<<<dim3(MROWS), w32, 0, stream>>>(
        bo, vis, ln2_g + l * D_, ln2_b + l * D_, nullptr, vis, 0);
  }

  // cosine similarity
  rownorm_kernel<<<dim3(MROWS), w32, 0, stream>>>(vis, rv);
  rownorm_kernel<<<dim3(TROWS), w32, 0, stream>>>(txt, rt);
  sim_kernel<<<dim3(S_ / 16, B_), w32, 0, stream>>>(vis, txt, rv, rt, simb);

  // optimal assignment (one wave per batch)
  lap_kernel<<<dim3(B_), w32, 0, stream>>>(simb, steps);
  gather_cost_kernel<<<dim3(B_), dim3(64), 0, stream>>>(simb, steps, simm, outp + B_);

  // fused head: vis part via WMMA (K=256 slice of fus_w, row stride 513)
  gemm_wmma_kernel<<<dim3(D_ / 64, MROWS / 64), w128, 0, stream>>>(
      vis, fus_w, fus_b, pre, D_, D_, 513, D_, 0);
  fus_add_kernel<<<dim3(N_, B_), dim3(256), 0, stream>>>(txt, fus_w, simm, steps, pre);
  pool_kernel<<<dim3(B_), dim3(256), 0, stream>>>(pre, pooled);
  cls_kernel<<<dim3(B_), dim3(64), 0, stream>>>(pooled, cls1_w, cls1_b, cls2_w, cls2_b, outp);
}
